// InvariantMultiACE_35399120453811
// MI455X (gfx1250) — compile-verified, matrix operand order checked by reading, not run on hardware
//
#include <hip/hip_runtime.h>
#include <hip/hip_bf16.h>

typedef __attribute__((ext_vector_type(16))) _Float16 v16h;
typedef __attribute__((ext_vector_type(8)))  _Float16 v8h;
typedef __attribute__((ext_vector_type(8)))  float    v8f;

#define PI_F 3.14159265358979323846f
#define RMAXF 5.0f

__constant__ int c_lidx[16] = {0,1,1,1,2,2,2,2,2,3,3,3,3,3,3,3};

// ---------------------------------------------------------------------------
// small elementwise kernels
// ---------------------------------------------------------------------------
__global__ void elem_k(const float* __restrict__ attrs, int* __restrict__ elem,
                       int N, int NE) {
  int n = blockIdx.x * blockDim.x + threadIdx.x;
  if (n >= N) return;
  int best = 0; float bv = attrs[(long)n * NE];
  for (int e = 1; e < NE; ++e) {
    float v = attrs[(long)n * NE + e];
    if (v > bv) { bv = v; best = e; }
  }
  elem[n] = best;
}

__global__ void embed_k(const float* __restrict__ W, const int* __restrict__ elem,
                        float* __restrict__ h, int N, int K) {
  long i = (long)blockIdx.x * blockDim.x + threadIdx.x;
  if (i >= (long)N * K) return;
  int n = (int)(i / K), k = (int)(i % K);
  h[i] = W[(long)elem[n] * K + k];
}

__global__ void gh_init_k(const float* __restrict__ Wro, float* __restrict__ gh,
                          int N, int K) {
  long i = (long)blockIdx.x * blockDim.x + threadIdx.x;
  if (i >= (long)N * K) return;
  gh[i] = Wro[(int)(i % K)];
}

__global__ void silu_fwd_k(const float* __restrict__ z, float* __restrict__ a, long n) {
  long i = (long)blockIdx.x * blockDim.x + threadIdx.x;
  if (i >= n) return;
  float zz = z[i];
  a[i] = zz / (1.f + __expf(-zz));
}

__global__ void silu_bwd_k(const float* __restrict__ z, const float* __restrict__ ga,
                           float* __restrict__ gz, long n) {
  long i = (long)blockIdx.x * blockDim.x + threadIdx.x;
  if (i >= n) return;
  float zz = z[i];
  float s = 1.f / (1.f + __expf(-zz));
  gz[i] = ga[i] * s * (1.f + zz * (1.f - s));
}

// pad W_r1 [2,8,64] -> [2,32,64] zero-filled
__global__ void pad_wr1_k(const float* __restrict__ Wr1, float* __restrict__ Wp) {
  int i = blockIdx.x * blockDim.x + threadIdx.x;
  if (i >= 2 * 32 * 64) return;
  int it = i / (32 * 64);
  int j  = (i / 64) & 31;
  int c  = i & 63;
  Wp[i] = (j < 8) ? Wr1[(long)it * 8 * 64 + j * 64 + c] : 0.f;
}

// U3 [16,16,16,8] f32 -> U3t [cp=128][ab=256] f16 (B^T layout for T3 GEMM)
__global__ void u3t_k(const float* __restrict__ U3, _Float16* __restrict__ U3t) {
  int i = blockIdx.x * blockDim.x + threadIdx.x;
  if (i >= 128 * 256) return;
  int cp = i >> 8;
  int ab = i & 255;
  int cidx = cp >> 3, p = cp & 7;
  U3t[(long)cp * 256 + ab] = (_Float16)U3[((long)ab * 16 + cidx) * 8 + p];
}

// U3 flat f32 -> f16 (natural [ab][cp] layout; B^T fragments for gAA GEMM)
__global__ void u3h_k(const float* __restrict__ U3, _Float16* __restrict__ U3h) {
  int i = blockIdx.x * blockDim.x + threadIdx.x;
  if (i >= 128 * 256) return;
  U3h[i] = (_Float16)U3[i];
}

// ---------------------------------------------------------------------------
// WMMA tile GEMM (full tiles only: M%16==0, Nn%16==0, Kd%32==0).
// ---------------------------------------------------------------------------
template <int TRANSB, int ACC>
__global__ void gemm_wmma(const float* __restrict__ A, int lda,
                          const float* __restrict__ B, int ldb,
                          float* __restrict__ C, int ldc,
                          int Kd, long aS, long bS, long cS, int bUseLidx) {
  int z = blockIdx.z;
  A += (long)z * aS;
  C += (long)z * cS;
  if (bUseLidx) B += (long)c_lidx[z] * bS;
  else          B += (long)z * bS;

  const int lane = threadIdx.x & 31;
  const int l16  = lane & 15;
  const int half = lane >> 4;
  const int m0 = blockIdx.y * 16;
  const int n0 = blockIdx.x * 16;

  const float* Arow = A + (long)(m0 + l16) * lda + 8 * half;
  const float* Bp;
  if (TRANSB) Bp = B + (long)(n0 + l16) * ldb + 8 * half;
  else        Bp = B + n0 + l16;
  float* Crow = C + (long)(m0 + 8 * half) * ldc + n0 + l16;

  v8f acc = {};
  if (ACC) {
#pragma unroll
    for (int r = 0; r < 8; ++r) acc[r] = Crow[(long)r * ldc];
  }

  for (int k0 = 0; k0 < Kd; k0 += 32) {
    if (k0 + 32 < Kd) {
      __builtin_prefetch(Arow + k0 + 32, 0, 1);
      if (TRANSB) __builtin_prefetch(Bp + k0 + 32, 0, 1);
    }
    float4 a0 = *(const float4*)(Arow + k0);
    float4 a1 = *(const float4*)(Arow + k0 + 4);
    float4 a2 = *(const float4*)(Arow + k0 + 16);
    float4 a3 = *(const float4*)(Arow + k0 + 20);
    v16h af, bf;
    af[0] = (_Float16)a0.x; af[1] = (_Float16)a0.y; af[2] = (_Float16)a0.z; af[3] = (_Float16)a0.w;
    af[4] = (_Float16)a1.x; af[5] = (_Float16)a1.y; af[6] = (_Float16)a1.z; af[7] = (_Float16)a1.w;
    af[8]  = (_Float16)a2.x; af[9]  = (_Float16)a2.y; af[10] = (_Float16)a2.z; af[11] = (_Float16)a2.w;
    af[12] = (_Float16)a3.x; af[13] = (_Float16)a3.y; af[14] = (_Float16)a3.z; af[15] = (_Float16)a3.w;
    if (TRANSB) {
      float4 b0 = *(const float4*)(Bp + k0);
      float4 b1 = *(const float4*)(Bp + k0 + 4);
      float4 b2 = *(const float4*)(Bp + k0 + 16);
      float4 b3 = *(const float4*)(Bp + k0 + 20);
      bf[0] = (_Float16)b0.x; bf[1] = (_Float16)b0.y; bf[2] = (_Float16)b0.z; bf[3] = (_Float16)b0.w;
      bf[4] = (_Float16)b1.x; bf[5] = (_Float16)b1.y; bf[6] = (_Float16)b1.z; bf[7] = (_Float16)b1.w;
      bf[8]  = (_Float16)b2.x; bf[9]  = (_Float16)b2.y; bf[10] = (_Float16)b2.z; bf[11] = (_Float16)b2.w;
      bf[12] = (_Float16)b3.x; bf[13] = (_Float16)b3.y; bf[14] = (_Float16)b3.z; bf[15] = (_Float16)b3.w;
    } else {
      const float* Bk = Bp + (long)(k0 + 8 * half) * ldb;
#pragma unroll
      for (int e = 0; e < 8; ++e)  bf[e] = (_Float16)Bk[(long)e * ldb];
      const float* Bk2 = Bk + (long)16 * ldb;
#pragma unroll
      for (int e = 0; e < 8; ++e)  bf[8 + e] = (_Float16)Bk2[(long)e * ldb];
    }
    acc = __builtin_amdgcn_wmma_f32_16x16x32_f16(false, af, false, bf,
                                                 (short)0, acc, false, false);
  }

#pragma unroll
  for (int r = 0; r < 8; ++r) Crow[(long)r * ldc] = acc[r];
}

// ---------------------------------------------------------------------------
// Edge geometry: Y (16 real sph harmonics) and K-padded radial basis Rp[E,32].
// ---------------------------------------------------------------------------
__global__ void edge_geom_k(const float* __restrict__ pos,
                            const float* __restrict__ shifts,
                            const int* __restrict__ ei, int E,
                            float* __restrict__ Y, float* __restrict__ Rp) {
  int e = blockIdx.x * blockDim.x + threadIdx.x;
  if (e >= E) return;
  int s = ei[e], r = ei[E + e];
  float vx = pos[(long)r * 3 + 0] - pos[(long)s * 3 + 0] + shifts[(long)e * 3 + 0];
  float vy = pos[(long)r * 3 + 1] - pos[(long)s * 3 + 1] + shifts[(long)e * 3 + 1];
  float vz = pos[(long)r * 3 + 2] - pos[(long)s * 3 + 2] + shifts[(long)e * 3 + 2];
  float rr = sqrtf(vx * vx + vy * vy + vz * vz + 1e-12f);
  float x = vx / rr, y = vy / rr, z = vz / rr;
  const float s3 = 1.7320508075688772f, s15 = 3.872983346207417f;
  const float s5 = 2.23606797749979f,  s70 = 8.366600265340756f;
  const float s105 = 10.246950765959598f, s42 = 6.48074069840786f;
  const float s7 = 2.6457513110645907f;
  float Yv[16];
  Yv[0]  = 1.f;
  Yv[1]  = s3 * x;  Yv[2] = s3 * y;  Yv[3] = s3 * z;
  Yv[4]  = s15 * x * y;  Yv[5] = s15 * y * z;
  Yv[6]  = 0.5f * s5 * (3.f * z * z - 1.f);
  Yv[7]  = s15 * x * z;
  Yv[8]  = 0.5f * s15 * (x * x - y * y);
  Yv[9]  = 0.25f * s70 * y * (3.f * x * x - y * y);
  Yv[10] = s105 * x * y * z;
  Yv[11] = 0.25f * s42 * y * (5.f * z * z - 1.f);
  Yv[12] = 0.5f * s7 * z * (5.f * z * z - 3.f);
  Yv[13] = 0.25f * s42 * x * (5.f * z * z - 1.f);
  Yv[14] = 0.5f * s105 * z * (x * x - y * y);
  Yv[15] = 0.25f * s70 * x * (3.f * x * x - y * y);
#pragma unroll
  for (int l = 0; l < 16; ++l) Y[(long)e * 16 + l] = Yv[l];

  float u = rr / RMAXF;
  float u2 = u * u, u4 = u2 * u2, u5 = u4 * u;
  float f = 1.f - 21.f * u5 + 35.f * u5 * u - 15.f * u5 * u2;
  if (u >= 1.f) f = 0.f;
  const float c = 0.6324555320336759f; // sqrt(2/5)
#pragma unroll
  for (int n = 1; n <= 8; ++n) {
    float arg = (float)n * PI_F * rr / RMAXF;
    Rp[(long)e * 32 + (n - 1)] = c * sinf(arg) / (rr + 1e-9f) * f;
  }
#pragma unroll
  for (int j = 8; j < 32; ++j) Rp[(long)e * 32 + j] = 0.f;
}

// ---------------------------------------------------------------------------
// Edge message forward / backward (atomic scatter over recv)
// ---------------------------------------------------------------------------
__global__ void msg_fwd_k(const float* __restrict__ w, const float* __restrict__ hup,
                          const float* __restrict__ Y, const int* __restrict__ ei,
                          float* __restrict__ agg, int E, int N, int K) {
  long gid = (long)blockIdx.x * blockDim.x + threadIdx.x;
  int edge = (int)(gid >> 5), lane = (int)(gid & 31);
  if (edge >= E) return;
  int s = ei[edge], r = ei[E + edge];
  float y[16];
#pragma unroll
  for (int l = 0; l < 16; ++l) y[l] = Y[(long)edge * 16 + l];
#pragma unroll
  for (int q = 0; q < 4; ++q) {
    int k = lane + 32 * q;
    float hv = hup[(long)s * K + k];
    float4 w4 = *(const float4*)&w[((long)edge * K + k) * 4];
    float wl[4] = {w4.x, w4.y, w4.z, w4.w};
#pragma unroll
    for (int lm = 0; lm < 16; ++lm) {
      float val = wl[c_lidx[lm]] * hv * y[lm] * 0.0625f;
      atomicAdd(&agg[((long)lm * N + r) * K + k], val);
    }
  }
}

__global__ void msg_bwd_k(const float* __restrict__ gagg, const float* __restrict__ w,
                          const float* __restrict__ hup, const float* __restrict__ Y,
                          const int* __restrict__ ei,
                          float* __restrict__ ghup, float* __restrict__ gw,
                          float* __restrict__ gY, int E, int N, int K) {
  long gid = (long)blockIdx.x * blockDim.x + threadIdx.x;
  int edge = (int)(gid >> 5), lane = (int)(gid & 31);
  if (edge >= E) return;
  int s = ei[edge], r = ei[E + edge];
  float y[16], gy[16];
#pragma unroll
  for (int l = 0; l < 16; ++l) { y[l] = Y[(long)edge * 16 + l]; gy[l] = 0.f; }
#pragma unroll
  for (int q = 0; q < 4; ++q) {
    int k = lane + 32 * q;
    float hv = hup[(long)s * K + k];
    float4 w4 = *(const float4*)&w[((long)edge * K + k) * 4];
    float wl[4] = {w4.x, w4.y, w4.z, w4.w};
    float ghv = 0.f, gwl[4] = {0.f, 0.f, 0.f, 0.f};
#pragma unroll
    for (int lm = 0; lm < 16; ++lm) {
      float gm = gagg[((long)lm * N + r) * K + k] * 0.0625f;
      int l = c_lidx[lm];
      ghv     += gm * wl[l] * y[lm];
      gwl[l]  += gm * hv * y[lm];
      gy[lm]  += gm * wl[l] * hv;
    }
    atomicAdd(&ghup[(long)s * K + k], ghv);
    float4 go; go.x = gwl[0]; go.y = gwl[1]; go.z = gwl[2]; go.w = gwl[3];
    *(float4*)&gw[((long)edge * K + k) * 4] = go;
  }
#pragma unroll
  for (int l = 0; l < 16; ++l) {
    float v = gy[l];
    for (int m = 16; m >= 1; m >>= 1) v += __shfl_xor(v, m);
    if (lane == 0) gY[(long)edge * 16 + l] += v;
  }
}

// ---------------------------------------------------------------------------
// Skip connection fwd/bwd
// ---------------------------------------------------------------------------
__global__ void skip_fwd_k(const float* __restrict__ h, const float* __restrict__ Wskip,
                           const int* __restrict__ elem, float* __restrict__ A0,
                           int N, int K) {
  __shared__ float sh[128];
  int n = blockIdx.x, k = threadIdx.x;
  sh[k] = h[(long)n * K + k];
  __syncthreads();
  const float* W = Wskip + (long)elem[n] * K * K;
  float acc = 0.f;
  for (int j = 0; j < K; ++j) acc += sh[j] * W[(long)j * K + k];
  A0[(long)n * K + k] += acc;
}

__global__ void skip_bwd_k(const float* __restrict__ gA0, const float* __restrict__ Wskip,
                           const int* __restrict__ elem, float* __restrict__ gh,
                           int N, int K) {
  __shared__ float sg[128];
  int n = blockIdx.x, j = threadIdx.x;
  sg[j] = gA0[(long)n * K + j];
  __syncthreads();
  const float* W = Wskip + (long)elem[n] * K * K;
  float acc = 0.f;
  for (int k = 0; k < K; ++k) acc += sg[k] * W[(long)j * K + k];
  gh[(long)n * K + j] += acc;
}

// ---------------------------------------------------------------------------
// Equivariant contraction FORWARD via WMMA (one workgroup / node, 8 waves).
// ---------------------------------------------------------------------------
__global__ void contract_fwd_wmma(const float* __restrict__ A,
                                  const float* __restrict__ U1, const float* __restrict__ U2,
                                  const _Float16* __restrict__ U3t,
                                  const float* __restrict__ Wc1, const float* __restrict__ Wc2,
                                  const float* __restrict__ Wc3,
                                  const int* __restrict__ elem,
                                  float* __restrict__ Bsum, int N, int K) {
  __shared__ float sa[128 * 17];
  __shared__ _Float16 AAh[64 * 272];
  __shared__ float sw3[8 * 128];
  __shared__ float sb3[128];
  __shared__ float sU2[16 * 16 * 4];
  __shared__ float sU1[32];

  const int n = blockIdx.x;
  const int tid = threadIdx.x;
  const int lane = tid & 31;
  const int l16  = lane & 15;
  const int half = lane >> 4;
  const int wid  = tid >> 5;
  const int e = elem[n];

  for (int idx = tid; idx < 128 * 16; idx += 256) {
    int l = idx >> 7, k = idx & 127;
    sa[k * 17 + l] = A[((long)l * N + n) * K + k];
  }
  for (int idx = tid; idx < 16 * 16 * 4; idx += 256) sU2[idx] = U2[idx];
  if (tid < 32) sU1[tid] = U1[tid];
  for (int idx = tid; idx < 8 * 128; idx += 256)
    sw3[idx] = Wc3[((long)e * 8 + (idx >> 7)) * K + (idx & 127)];
  if (tid < 128) sb3[tid] = 0.f;
  __syncthreads();

  float b12 = 0.f;
  if (tid < 128) {
    const int k = tid;
    const float* a = &sa[k * 17];
    float w1[2], w2[4];
#pragma unroll
    for (int p = 0; p < 2; ++p) w1[p] = Wc1[((long)e * 2 + p) * K + k];
#pragma unroll
    for (int p = 0; p < 4; ++p) w2[p] = Wc2[((long)e * 4 + p) * K + k];
#pragma unroll 1
    for (int x = 0; x < 16; ++x) {
      float c1 = sU1[x * 2] * w1[0] + sU1[x * 2 + 1] * w1[1];
      b12 += a[x] * c1;
    }
#pragma unroll 1
    for (int x = 0; x < 16; ++x) {
      float ax = a[x];
#pragma unroll 1
      for (int y = 0; y < 16; ++y) {
        const float* u2 = &sU2[(x * 16 + y) * 4];
        float c2 = u2[0] * w2[0] + u2[1] * w2[1] + u2[2] * w2[2] + u2[3] * w2[3];
        b12 += ax * a[y] * c2;
      }
    }
  }

  const int rt = wid & 3;
  const int ctbase = (wid >> 2) * 4;
  for (int half_k = 0; half_k < 2; ++half_k) {
    __syncthreads();
    for (int idx = tid; idx < 64 * 256; idx += 256) {
      int row = idx >> 8;
      int ab  = idx & 255;
      int kg  = half_k * 64 + row;
      AAh[row * 272 + ab] = (_Float16)(sa[kg * 17 + (ab >> 4)] * sa[kg * 17 + (ab & 15)]);
    }
    __syncthreads();
#pragma unroll 1
    for (int j = 0; j < 4; ++j) {
      const int ct = ctbase + j;
      const _Float16* Ar = &AAh[(rt * 16 + l16) * 272 + 8 * half];
      const _Float16* Br = U3t + (long)(ct * 16 + l16) * 256 + 8 * half;
      v8f acc = {};
#pragma unroll
      for (int k0 = 0; k0 < 256; k0 += 32) {
        v8h alo = *(const v8h*)(Ar + k0);
        v8h ahi = *(const v8h*)(Ar + k0 + 16);
        v8h blo = *(const v8h*)(Br + k0);
        v8h bhi = *(const v8h*)(Br + k0 + 16);
        v16h af = __builtin_shufflevector(alo, ahi, 0,1,2,3,4,5,6,7,8,9,10,11,12,13,14,15);
        v16h bf = __builtin_shufflevector(blo, bhi, 0,1,2,3,4,5,6,7,8,9,10,11,12,13,14,15);
        acc = __builtin_amdgcn_wmma_f32_16x16x32_f16(false, af, false, bf,
                                                     (short)0, acc, false, false);
      }
#pragma unroll
      for (int r = 0; r < 8; ++r) {
        int kg  = half_k * 64 + rt * 16 + r + 8 * half;
        int col = ct * 16 + l16;
        float v = acc[r] * sa[kg * 17 + (col >> 3)] * sw3[(col & 7) * 128 + kg];
        v += __shfl_xor(v, 1);
        v += __shfl_xor(v, 2);
        v += __shfl_xor(v, 4);
        v += __shfl_xor(v, 8);
        if (l16 == 0) atomicAdd(&sb3[kg], v);
      }
    }
  }
  __syncthreads();
  if (tid < 128) Bsum[(long)n * K + tid] = b12 + sb3[tid];
}

// ---------------------------------------------------------------------------
// Equivariant contraction BACKWARD via WMMA (one workgroup / node, 8 waves).
// GEMM1: T3 = AA @ U3r      -> ga[c] += sum_p T3[k,c,p]*w3[p,k]
// GEMM2: gAA = V @ U3r^T    (V[k,cp]=a[c]*w3[p,k]) -> ga[x]+=gAA*a[y] (+sym)
// b1/b2 adjoints scalar in an exclusive phase; final gA = gb * ga.
// ---------------------------------------------------------------------------
__global__ void contract_bwd_wmma(const float* __restrict__ A, const float* __restrict__ gb,
                                  const float* __restrict__ U1, const float* __restrict__ U2,
                                  const _Float16* __restrict__ U3t,
                                  const _Float16* __restrict__ U3h,
                                  const float* __restrict__ Wc1, const float* __restrict__ Wc2,
                                  const float* __restrict__ Wc3,
                                  const int* __restrict__ elem,
                                  float* __restrict__ gA, int N, int K) {
  __shared__ float sa[128 * 17];
  __shared__ float sga[128 * 17];
  __shared__ _Float16 AAh[64 * 272];   // reused as Vh (stride 136)
  __shared__ float sw3[8 * 128];
  __shared__ float sgb[128];
  __shared__ float sU2[16 * 16 * 4];
  __shared__ float sU1[32];

  const int n = blockIdx.x;
  const int tid = threadIdx.x;
  const int lane = tid & 31;
  const int l16  = lane & 15;
  const int half = lane >> 4;
  const int wid  = tid >> 5;
  const int e = elem[n];

  for (int idx = tid; idx < 128 * 16; idx += 256) {
    int l = idx >> 7, k = idx & 127;
    sa[k * 17 + l] = A[((long)l * N + n) * K + k];
    sga[k * 17 + l] = 0.f;
  }
  for (int idx = tid; idx < 16 * 16 * 4; idx += 256) sU2[idx] = U2[idx];
  if (tid < 32) sU1[tid] = U1[tid];
  for (int idx = tid; idx < 8 * 128; idx += 256)
    sw3[idx] = Wc3[((long)e * 8 + (idx >> 7)) * K + (idx & 127)];
  if (tid < 128) sgb[tid] = gb[(long)n * K + tid];
  __syncthreads();

  // scalar b1 + b2 adjoint (exclusive row access; before any atomics)
  if (tid < 128) {
    const int k = tid;
    float* a  = &sa[k * 17];
    float* ga = &sga[k * 17];
    float w1[2], w2[4];
#pragma unroll
    for (int p = 0; p < 2; ++p) w1[p] = Wc1[((long)e * 2 + p) * K + k];
#pragma unroll
    for (int p = 0; p < 4; ++p) w2[p] = Wc2[((long)e * 4 + p) * K + k];
#pragma unroll 1
    for (int x = 0; x < 16; ++x)
      ga[x] += sU1[x * 2] * w1[0] + sU1[x * 2 + 1] * w1[1];
#pragma unroll 1
    for (int x = 0; x < 16; ++x) {
      float ax = a[x];
#pragma unroll 1
      for (int y = 0; y < 16; ++y) {
        const float* u2 = &sU2[(x * 16 + y) * 4];
        float c2 = u2[0] * w2[0] + u2[1] * w2[1] + u2[2] * w2[2] + u2[3] * w2[3];
        ga[x] += c2 * a[y];
        ga[y] += c2 * ax;
      }
    }
  }

  const int rt = wid & 3;
  const int ctbase = (wid >> 2) * 4;

  // ---- GEMM1: T3 tiles; epilogue reduces over p (8 lanes) ----
  for (int half_k = 0; half_k < 2; ++half_k) {
    __syncthreads();
    for (int idx = tid; idx < 64 * 256; idx += 256) {
      int row = idx >> 8;
      int ab  = idx & 255;
      int kg  = half_k * 64 + row;
      AAh[row * 272 + ab] = (_Float16)(sa[kg * 17 + (ab >> 4)] * sa[kg * 17 + (ab & 15)]);
    }
    __syncthreads();
#pragma unroll 1
    for (int j = 0; j < 4; ++j) {
      const int ct = ctbase + j;
      const _Float16* Ar = &AAh[(rt * 16 + l16) * 272 + 8 * half];
      const _Float16* Br = U3t + (long)(ct * 16 + l16) * 256 + 8 * half;
      v8f acc = {};
#pragma unroll
      for (int k0 = 0; k0 < 256; k0 += 32) {
        v8h alo = *(const v8h*)(Ar + k0);
        v8h ahi = *(const v8h*)(Ar + k0 + 16);
        v8h blo = *(const v8h*)(Br + k0);
        v8h bhi = *(const v8h*)(Br + k0 + 16);
        v16h af = __builtin_shufflevector(alo, ahi, 0,1,2,3,4,5,6,7,8,9,10,11,12,13,14,15);
        v16h bf = __builtin_shufflevector(blo, bhi, 0,1,2,3,4,5,6,7,8,9,10,11,12,13,14,15);
        acc = __builtin_amdgcn_wmma_f32_16x16x32_f16(false, af, false, bf,
                                                     (short)0, acc, false, false);
      }
#pragma unroll
      for (int r = 0; r < 8; ++r) {
        int kg  = half_k * 64 + rt * 16 + r + 8 * half;
        int col = ct * 16 + l16;
        float v = acc[r] * sw3[(col & 7) * 128 + kg];
        v += __shfl_xor(v, 1);
        v += __shfl_xor(v, 2);
        v += __shfl_xor(v, 4);
        if ((l16 & 7) == 0) atomicAdd(&sga[kg * 17 + (col >> 3)], v);
      }
    }
  }

  // ---- GEMM2: gAA = V @ U3r^T ; scatter into ga with symmetry ----
  _Float16* Vh = AAh;                  // reuse, stride 136
  const int ct2base = (wid >> 2) * 8;
  for (int half_k = 0; half_k < 2; ++half_k) {
    __syncthreads();
    for (int idx = tid; idx < 64 * 128; idx += 256) {
      int row = idx >> 7;
      int cp  = idx & 127;
      int kg  = half_k * 64 + row;
      Vh[row * 136 + cp] = (_Float16)(sa[kg * 17 + (cp >> 3)] * sw3[(cp & 7) * 128 + kg]);
    }
    __syncthreads();
#pragma unroll 1
    for (int j = 0; j < 8; ++j) {
      const int ct = ct2base + j;
      const _Float16* Ar = &Vh[(rt * 16 + l16) * 136 + 8 * half];
      const _Float16* Br = U3h + (long)(ct * 16 + l16) * 128 + 8 * half;
      v8f acc = {};
#pragma unroll
      for (int k0 = 0; k0 < 128; k0 += 32) {
        v8h alo = *(const v8h*)(Ar + k0);
        v8h ahi = *(const v8h*)(Ar + k0 + 16);
        v8h blo = *(const v8h*)(Br + k0);
        v8h bhi = *(const v8h*)(Br + k0 + 16);
        v16h af = __builtin_shufflevector(alo, ahi, 0,1,2,3,4,5,6,7,8,9,10,11,12,13,14,15);
        v16h bf = __builtin_shufflevector(blo, bhi, 0,1,2,3,4,5,6,7,8,9,10,11,12,13,14,15);
        acc = __builtin_amdgcn_wmma_f32_16x16x32_f16(false, af, false, bf,
                                                     (short)0, acc, false, false);
      }
#pragma unroll
      for (int r = 0; r < 8; ++r) {
        int kg = half_k * 64 + rt * 16 + r + 8 * half;
        int ab = ct * 16 + l16;
        int x = ab >> 4, y = ab & 15;
        float v = acc[r];
        atomicAdd(&sga[kg * 17 + x], v * sa[kg * 17 + y]);
        atomicAdd(&sga[kg * 17 + y], v * sa[kg * 17 + x]);
      }
    }
  }
  __syncthreads();
  if (tid < 128) {
    float g = sgb[tid];
#pragma unroll
    for (int l = 0; l < 16; ++l)
      gA[((long)l * N + n) * K + tid] = g * sga[tid * 17 + l];
  }
}

// ---------------------------------------------------------------------------
// Readout: energy per graph = sum over nodes of (h2 . Wro + E0[elem]).
// ---------------------------------------------------------------------------
__global__ void readout_k(const float* __restrict__ h2, const float* __restrict__ Wro,
                          const int* __restrict__ elem, const float* __restrict__ ae,
                          const int* __restrict__ batch, float* __restrict__ out,
                          int N, int K) {
  int n = blockIdx.x * blockDim.x + threadIdx.x;
  if (n >= N) return;
  float s = 0.f;
  for (int k = 0; k < K; ++k) s += h2[(long)n * K + k] * Wro[k];
  s += ae[elem[n]];
  atomicAdd(&out[batch[n]], s);
}

// ---------------------------------------------------------------------------
// Geometry backward: gY, gRp (stride 16) -> dE/dvec -> forces (= -grad).
// ---------------------------------------------------------------------------
__global__ void geom_bwd_k(const float* __restrict__ pos, const float* __restrict__ shifts,
                           const int* __restrict__ ei,
                           const float* __restrict__ gY, const float* __restrict__ gRp,
                           float* __restrict__ F, int E) {
  int e = blockIdx.x * blockDim.x + threadIdx.x;
  if (e >= E) return;
  int s = ei[e], r = ei[E + e];
  float vx = pos[(long)r * 3 + 0] - pos[(long)s * 3 + 0] + shifts[(long)e * 3 + 0];
  float vy = pos[(long)r * 3 + 1] - pos[(long)s * 3 + 1] + shifts[(long)e * 3 + 1];
  float vz = pos[(long)r * 3 + 2] - pos[(long)s * 3 + 2] + shifts[(long)e * 3 + 2];
  float rr = sqrtf(vx * vx + vy * vy + vz * vz + 1e-12f);
  float x = vx / rr, y = vy / rr, z = vz / rr;
  const float s3 = 1.7320508075688772f, s15 = 3.872983346207417f;
  const float s5 = 2.23606797749979f,  s70 = 8.366600265340756f;
  const float s105 = 10.246950765959598f, s42 = 6.48074069840786f;
  const float s7 = 2.6457513110645907f;
  float gy[16];
#pragma unroll
  for (int l = 0; l < 16; ++l) gy[l] = gY[(long)e * 16 + l];
  float gx = 0.f, gyy = 0.f, gz = 0.f;
  gx  += gy[1] * s3;  gyy += gy[2] * s3;  gz += gy[3] * s3;
  gx  += gy[4] * s15 * y;  gyy += gy[4] * s15 * x;
  gyy += gy[5] * s15 * z;  gz  += gy[5] * s15 * y;
  gz  += gy[6] * 3.f * s5 * z;
  gx  += gy[7] * s15 * z;  gz  += gy[7] * s15 * x;
  gx  += gy[8] * s15 * x;  gyy -= gy[8] * s15 * y;
  gx  += gy[9] * 1.5f * s70 * x * y;
  gyy += gy[9] * 0.75f * s70 * (x * x - y * y);
  gx  += gy[10] * s105 * y * z;  gyy += gy[10] * s105 * x * z;  gz += gy[10] * s105 * x * y;
  gyy += gy[11] * 0.25f * s42 * (5.f * z * z - 1.f);
  gz  += gy[11] * 2.5f * s42 * y * z;
  gz  += gy[12] * 0.5f * s7 * (15.f * z * z - 3.f);
  gx  += gy[13] * 0.25f * s42 * (5.f * z * z - 1.f);
  gz  += gy[13] * 2.5f * s42 * x * z;
  gx  += gy[14] * s105 * x * z;  gyy -= gy[14] * s105 * y * z;
  gz  += gy[14] * 0.5f * s105 * (x * x - y * y);
  gx  += gy[15] * 0.25f * s70 * (9.f * x * x - y * y);
  gyy -= gy[15] * 0.5f * s70 * x * y;
  float gd = gx * x + gyy * y + gz * z;
  float gvx = (gx - x * gd) / rr;
  float gvy = (gyy - y * gd) / rr;
  float gvz = (gz - z * gd) / rr;
  float u = rr / RMAXF;
  float u2 = u * u, u4 = u2 * u2, u5 = u4 * u;
  float f = 1.f - 21.f * u5 + 35.f * u5 * u - 15.f * u5 * u2;
  float dfdr = (-105.f * u4 + 210.f * u5 - 105.f * u5 * u) / RMAXF;
  if (u >= 1.f) { f = 0.f; dfdr = 0.f; }
  const float c = 0.6324555320336759f;
  float inv = 1.f / (rr + 1e-9f);
  float gr = 0.f;
#pragma unroll
  for (int n = 1; n <= 8; ++n) {
    float arg = (float)n * PI_F * rr / RMAXF;
    float sn = sinf(arg), cs = cosf(arg);
    float b  = c * sn * inv;
    float db = c * ((float)n * PI_F / RMAXF) * cs * inv - c * sn * inv * inv;
    gr += gRp[(long)e * 16 + (n - 1)] * (db * f + b * dfdr);
  }
  gvx += gr * x;  gvy += gr * y;  gvz += gr * z;
  atomicAdd(&F[(long)r * 3 + 0], -gvx);
  atomicAdd(&F[(long)r * 3 + 1], -gvy);
  atomicAdd(&F[(long)r * 3 + 2], -gvz);
  atomicAdd(&F[(long)s * 3 + 0], gvx);
  atomicAdd(&F[(long)s * 3 + 1], gvy);
  atomicAdd(&F[(long)s * 3 + 2], gvz);
}

// ---------------------------------------------------------------------------
// Host orchestration
// ---------------------------------------------------------------------------
extern "C" void kernel_launch(void* const* d_in, const int* in_sizes, int n_in,
                              void* d_out, int out_size, void* d_ws, size_t ws_size,
                              hipStream_t stream) {
  const float* node_attrs = (const float*)d_in[0];
  const float* positions  = (const float*)d_in[1];
  const float* shifts     = (const float*)d_in[2];
  const float* atomic_e   = (const float*)d_in[3];
  const float* W_embed    = (const float*)d_in[4];
  const float* W_up       = (const float*)d_in[5];
  const float* W_r1       = (const float*)d_in[6];
  const float* W_r2       = (const float*)d_in[7];
  const float* W_r3       = (const float*)d_in[8];
  const float* W_lin      = (const float*)d_in[9];
  const float* W_skip     = (const float*)d_in[10];
  const float* U1         = (const float*)d_in[11];
  const float* U2         = (const float*)d_in[12];
  const float* U3         = (const float*)d_in[13];
  const float* Wc1        = (const float*)d_in[14];
  const float* Wc2        = (const float*)d_in[15];
  const float* Wc3        = (const float*)d_in[16];
  const float* W_prod     = (const float*)d_in[17];
  const float* W_ro       = (const float*)d_in[18];
  const int*   ei         = (const int*)d_in[19];
  const int*   batch      = (const int*)d_in[20];
  (void)node_attrs; (void)n_in; (void)ws_size;

  const int N  = in_sizes[1] / 3;
  const int E  = in_sizes[2] / 3;
  const int NE = in_sizes[3];
  const int K  = 128, LM = 16, HID = 64, W4 = 4;
  const int nG = out_size - 3 * N;

  float* outE = (float*)d_out;
  float* outF = outE + nG;

  // ---- workspace carving ----
  char* base = (char*)d_ws;
  size_t off = 0;
  auto af = [&](size_t nf) -> float* {
    float* p = (float*)(base + off);
    off = (off + nf * sizeof(float) + 255) & ~(size_t)255;
    return p;
  };
  int*   elem   = (int*)af((size_t)N);
  float* h[3]   = { af((size_t)N * K), af((size_t)N * K), af((size_t)N * K) };
  float* Ysh    = af((size_t)E * LM);
  float* Rp     = af((size_t)E * 32);
  float* Wr1p   = af((size_t)2 * 32 * HID);
  _Float16* U3t = (_Float16*)af((size_t)128 * 256 / 2);
  _Float16* U3h = (_Float16*)af((size_t)128 * 256 / 2);
  float* hup[2] = { af((size_t)N * K), af((size_t)N * K) };
  float* z1[2]  = { af((size_t)E * HID), af((size_t)E * HID) };
  float* z2[2]  = { af((size_t)E * HID), af((size_t)E * HID) };
  float* a1     = af((size_t)E * HID);
  float* a2     = af((size_t)E * HID);
  float* wbuf[2]= { af((size_t)E * K * W4), af((size_t)E * K * W4) };
  float* agg    = af((size_t)LM * N * K);
  float* Amat[2]= { af((size_t)LM * N * K), af((size_t)LM * N * K) };
  float* Bsum   = af((size_t)N * K);
  float* gh     = af((size_t)N * K);
  float* ghn    = af((size_t)N * K);
  float* gb     = af((size_t)N * K);
  float* ghup   = af((size_t)N * K);
  float* gA     = af((size_t)LM * N * K);
  float* gagg   = af((size_t)LM * N * K);
  float* gw     = af((size_t)E * K * W4);
  float* ge1    = af((size_t)E * HID);
  float* ge2    = af((size_t)E * HID);
  float* gY     = af((size_t)E * LM);
  float* gRp    = af((size_t)E * 16);

  auto gemm = [&](const float* A, int lda, const float* B, int ldb, float* C, int ldc,
                  int M, int Nn, int Kd, int transB, int accum,
                  int nbatch, long aS, long bS, long cS, int useLidx) {
    dim3 g((Nn + 15) / 16, (M + 15) / 16, nbatch);
    if (!transB && !accum)
      gemm_wmma<0, 0><<<g, dim3(32), 0, stream>>>(A, lda, B, ldb, C, ldc, Kd, aS, bS, cS, useLidx);
    else if (transB && !accum)
      gemm_wmma<1, 0><<<g, dim3(32), 0, stream>>>(A, lda, B, ldb, C, ldc, Kd, aS, bS, cS, useLidx);
    else
      gemm_wmma<1, 1><<<g, dim3(32), 0, stream>>>(A, lda, B, ldb, C, ldc, Kd, aS, bS, cS, useLidx);
  };
  auto ew = [&](long n) { return dim3((unsigned)((n + 255) / 256)); };

  // ---- forward ----
  hipMemsetAsync(d_out, 0, (size_t)out_size * sizeof(float), stream);
  elem_k<<<ew(N), 256, 0, stream>>>(node_attrs, elem, N, NE);
  embed_k<<<ew((long)N * K), 256, 0, stream>>>(W_embed, elem, h[0], N, K);
  edge_geom_k<<<ew(E), 256, 0, stream>>>(positions, shifts, ei, E, Ysh, Rp);
  pad_wr1_k<<<ew(2 * 32 * HID), 256, 0, stream>>>(W_r1, Wr1p);
  u3t_k<<<ew(128 * 256), 256, 0, stream>>>(U3, U3t);
  u3h_k<<<ew(128 * 256), 256, 0, stream>>>(U3, U3h);

  for (int it = 0; it < 2; ++it) {
    gemm(h[it], K, W_up + (long)it * K * K, K, hup[it], K, N, K, K, 0, 0, 1, 0, 0, 0, 0);
    gemm(Rp, 32, Wr1p + (long)it * 32 * HID, HID, z1[it], HID, E, HID, 32, 0, 0, 1, 0, 0, 0, 0);
    silu_fwd_k<<<ew((long)E * HID), 256, 0, stream>>>(z1[it], a1, (long)E * HID);
    gemm(a1, HID, W_r2 + (long)it * HID * HID, HID, z2[it], HID, E, HID, HID, 0, 0, 1, 0, 0, 0, 0);
    silu_fwd_k<<<ew((long)E * HID), 256, 0, stream>>>(z2[it], a2, (long)E * HID);
    gemm(a2, HID, W_r3 + (long)it * HID * K * W4, K * W4, wbuf[it], K * W4,
         E, K * W4, HID, 0, 0, 1, 0, 0, 0, 0);
    hipMemsetAsync(agg, 0, (size_t)LM * N * K * sizeof(float), stream);
    msg_fwd_k<<<ew((long)E * 32), 256, 0, stream>>>(wbuf[it], hup[it], Ysh, ei, agg, E, N, K);
    gemm(agg, K, W_lin + (long)it * 4 * K * K, K, Amat[it], K, N, K, K, 0, 0,
         LM, (long)N * K, (long)K * K, (long)N * K, 1);
    skip_fwd_k<<<N, K, 0, stream>>>(h[it], W_skip + (long)it * NE * K * K, elem, Amat[it], N, K);
    contract_fwd_wmma<<<N, 256, 0, stream>>>(Amat[it], U1, U2, U3t,
                                             Wc1 + (long)it * NE * 2 * K,
                                             Wc2 + (long)it * NE * 4 * K,
                                             Wc3 + (long)it * NE * 8 * K,
                                             elem, Bsum, N, K);
    gemm(Bsum, K, W_prod + (long)it * K * K, K, h[it + 1], K, N, K, K, 0, 0, 1, 0, 0, 0, 0);
  }
  readout_k<<<ew(N), 256, 0, stream>>>(h[2], W_ro, elem, atomic_e, batch, outE, N, K);

  // ---- backward (forces) ----
  hipMemsetAsync(gY, 0, (size_t)E * LM * sizeof(float), stream);
  hipMemsetAsync(gRp, 0, (size_t)E * 16 * sizeof(float), stream);
  gh_init_k<<<ew((long)N * K), 256, 0, stream>>>(W_ro, gh, N, K);

  for (int it = 1; it >= 0; --it) {
    gemm(gh, K, W_prod + (long)it * K * K, K, gb, K, N, K, K, 1, 0, 1, 0, 0, 0, 0);
    contract_bwd_wmma<<<N, 256, 0, stream>>>(Amat[it], gb, U1, U2, U3t, U3h,
                                             Wc1 + (long)it * NE * 2 * K,
                                             Wc2 + (long)it * NE * 4 * K,
                                             Wc3 + (long)it * NE * 8 * K,
                                             elem, gA, N, K);
    hipMemsetAsync(ghn, 0, (size_t)N * K * sizeof(float), stream);
    skip_bwd_k<<<N, K, 0, stream>>>(gA, W_skip + (long)it * NE * K * K, elem, ghn, N, K);
    gemm(gA, K, W_lin + (long)it * 4 * K * K, K, gagg, K, N, K, K, 1, 0,
         LM, (long)N * K, (long)K * K, (long)N * K, 1);
    hipMemsetAsync(ghup, 0, (size_t)N * K * sizeof(float), stream);
    msg_bwd_k<<<ew((long)E * 32), 256, 0, stream>>>(gagg, wbuf[it], hup[it], Ysh, ei,
                                                    ghup, gw, gY, E, N, K);
    gemm(ghup, K, W_up + (long)it * K * K, K, ghn, K, N, K, K, 1, 1, 1, 0, 0, 0, 0);
    gemm(gw, K * W4, W_r3 + (long)it * HID * K * W4, K * W4, ge2, HID,
         E, HID, K * W4, 1, 0, 1, 0, 0, 0, 0);
    silu_bwd_k<<<ew((long)E * HID), 256, 0, stream>>>(z2[it], ge2, ge2, (long)E * HID);
    gemm(ge2, HID, W_r2 + (long)it * HID * HID, HID, ge1, HID, E, HID, HID, 1, 0, 1, 0, 0, 0, 0);
    silu_bwd_k<<<ew((long)E * HID), 256, 0, stream>>>(z1[it], ge1, ge1, (long)E * HID);
    gemm(ge1, HID, Wr1p + (long)it * 32 * HID, HID, gRp, 16, E, 16, HID, 1, 1, 1, 0, 0, 0, 0);
    float* t = gh; gh = ghn; ghn = t;
  }
  geom_bwd_k<<<ew(E), 256, 0, stream>>>(positions, shifts, ei, gY, gRp, outF, E);
}
